// QRNN3DDecoder_59124519796891
// MI455X (gfx1250) — matrix operand order repaired and checked by reading, not compiled, not run on hardware
//
#include <hip/hip_runtime.h>
#include <hip/hip_bf16.h>

typedef __bf16 bf16;
typedef __attribute__((ext_vector_type(16))) __bf16 v16bf;
typedef __attribute__((ext_vector_type(8)))  float  v8f;

// ---------------------------------------------------------------------------
// Weight prep: pack conv weights into WMMA A-fragment order (bf16).
// Fragment index = (ob*nCblk + cb)*27 + tap ; 32 lanes x 16 bf16 per fragment.
// A order used here (contiguous k): lane<16 holds row M=lane, elems e0..15 -> K0..15 ;
// lane>=16 holds row M=lane-16, e0..15 -> K16..31 (matches B staging).
// transpose=1 implements deconv: We[o][i][t]=w[i][o][26-t].
// ---------------------------------------------------------------------------
__global__ void wprep_kernel(const float* __restrict__ w, bf16* __restrict__ dst,
                             int Oeff, int Ieff, int nCblk, int transpose, int total)
{
    int i = blockIdx.x * blockDim.x + threadIdx.x;
    if (i >= total) return;                       // total = nOblk*nCblk*27*32
    int lane = i & 31;
    int frag = i >> 5;
    int tap  = frag % 27;
    int t    = frag / 27;
    int cb   = t % nCblk;
    int ob   = t / nCblk;
    int half = lane >> 4;
    int o    = ob * 16 + (lane & 15);
#pragma unroll
    for (int e = 0; e < 16; ++e) {
        int cblkc = half * 16 + e;
        int cin   = cb * 32 + cblkc;
        float v = 0.f;
        if (cin < Ieff)
            v = transpose ? w[((size_t)cin * Oeff + o) * 27 + (26 - tap)]
                          : w[((size_t)o * Ieff + cin) * 27 + tap];
        dst[(size_t)frag * 512 + lane * 16 + e] = (bf16)v;
    }
}

// ---------------------------------------------------------------------------
// Input prep: (A [+ B]) -> optional 2x nearest upsample -> bf16, packed as
// [cblk][n][Dp][Hp][Wp][32] with zero halo (Dp=D+2 etc) and zero channel pad.
// One thread per padded (n,dp,yp,xp) position; loops channels.
// ---------------------------------------------------------------------------
__global__ void inprep_kernel(const float* __restrict__ A, const float* __restrict__ B,
                              bf16* __restrict__ dst, int N, int Csrc, int Cpad,
                              int D, int H, int W, int up, int total)
{
    int i = blockIdx.x * blockDim.x + threadIdx.x;
    if (i >= total) return;                       // total = N*Dp*Hp*Wp
    int Dp = D + 2, Hp = H + 2, Wp = W + 2;
    int xp = i % Wp; int t = i / Wp;
    int yp = t % Hp; t /= Hp;
    int dp = t % Dp; int n = t / Dp;
    bool border = (dp == 0) | (dp == Dp - 1) | (yp == 0) | (yp == Hp - 1) |
                  (xp == 0) | (xp == Wp - 1);
    int sH = up ? (H >> 1) : H;
    int sW = up ? (W >> 1) : W;
    int d = dp - 1, y = yp - 1, x = xp - 1;
    int sy = up ? (y >> 1) : y;
    int sx = up ? (x >> 1) : x;
    size_t obase = ((((size_t)n * Dp + dp) * Hp + yp) * (size_t)Wp + xp) * 32;
    size_t cblkStride = (size_t)N * Dp * Hp * Wp * 32;
    for (int c = 0; c < Cpad; ++c) {
        float v = 0.f;
        if (!border && c < Csrc) {
            size_t si = ((((size_t)n * Csrc + c) * D + d) * (size_t)sH + sy) * sW + sx;
            v = A[si];
            if (B) v += B[si];
        }
        dst[(size_t)(c >> 5) * cblkStride + obase + (c & 31)] = (bf16)v;
    }
}

// ---------------------------------------------------------------------------
// Implicit-GEMM 3x3x3 conv via v_wmma_f32_16x16x32_bf16, fused BN + tanh/sigm.
// One wave computes a 32-outchannel x 16-xposition tile at fixed (n,d,y):
// two accumulators share every B fragment (halves input traffic, and the
// paired WMMAs are independent -> no D->A/B hazard stalls between them).
// Layer dims are template constants so the tile decode is shift/mul only.
// ---------------------------------------------------------------------------
template <int D, int H, int W, int Cout, int hidden, int nCblk>
__global__ __launch_bounds__(256)
void conv_wmma_kernel(const bf16* __restrict__ Bin, const bf16* __restrict__ Wfrag,
                      const float* __restrict__ bn, float* __restrict__ g,
                      int N, int totalWaves)
{
    int wid = blockIdx.x * 8 + ((int)threadIdx.x >> 5);
    if (wid >= totalWaves) return;                // wave-uniform; EXEC stays full
    int lane = threadIdx.x & 31;
    int lx   = lane & 15;
    int half = lane >> 4;

    constexpr int Wt  = W >> 4;
    constexpr int nOb = Cout >> 5;                // 32-channel tiles per wave
    constexpr int Dp = D + 2, Hp = H + 2, Wp = W + 2;

    int t   = wid;
    int xt  = t % Wt;  t /= Wt;
    int y   = t % H;   t /= H;
    int d   = t % D;   t /= D;
    int obp = t % nOb; t /= nOb;
    int n   = t;
    int x0  = xt << 4;

    v8f acc0 = {};
    v8f acc1 = {};
    for (int cb = 0; cb < nCblk; ++cb) {
        const bf16* wf0 = Wfrag + ((size_t)((obp * 2) * nCblk + cb) * 27) * 512
                        + lane * 16;
        const bf16* wf1 = wf0 + (size_t)nCblk * 27 * 512;
        size_t base = (((((size_t)cb * N + n) * Dp + d) * Hp + y) * (size_t)Wp
                       + (x0 + lx)) * 32 + half * 16;
#pragma unroll
        for (int dz = 0; dz < 3; ++dz)
#pragma unroll
            for (int dy = 0; dy < 3; ++dy)
#pragma unroll
                for (int dx = 0; dx < 3; ++dx) {
                    v16bf b = *reinterpret_cast<const v16bf*>(
                        Bin + base + ((size_t)(dz * Hp + dy) * Wp + dx) * 32);
                    v16bf a0 = *reinterpret_cast<const v16bf*>(wf0);
                    v16bf a1 = *reinterpret_cast<const v16bf*>(wf1);
                    acc0 = __builtin_amdgcn_wmma_f32_16x16x32_bf16(
                        false, a0, false, b, (short)0, acc0, false, false);
                    acc1 = __builtin_amdgcn_wmma_f32_16x16x32_bf16(
                        false, a1, false, b, (short)0, acc1, false, false);
                    wf0 += 512;
                    wf1 += 512;
                }
    }

    // Epilogue: C layout -> VGPR r: lane<16 => M=r, lane>=16 => M=r+8; N=lane&15
    constexpr size_t HW = (size_t)H * W;
    size_t obase = (size_t)y * W + x0 + lx;
#pragma unroll
    for (int r = 0; r < 8; ++r) {
#pragma unroll
        for (int q = 0; q < 2; ++q) {
            int   c = obp * 32 + q * 16 + half * 8 + r;
            float a = (q == 0) ? acc0[r] : acc1[r];
            float sc = bn[0 * Cout + c] * rsqrtf(bn[3 * Cout + c] + 1e-5f);
            float sh = bn[1 * Cout + c] - bn[2 * Cout + c] * sc;
            float v  = a * sc + sh;
            v = (c < hidden) ? tanhf(v) : (1.f / (1.f + expf(-v)));
            g[(((size_t)n * Cout + c) * D + d) * HW + obase] = v;
        }
    }
}

// ---------------------------------------------------------------------------
// QRNN scan: h[d] = f*h[d-1] + (1-f)*z, forward or reverse over D.
// Branchless direction via signed stride.
// ---------------------------------------------------------------------------
__global__ void scan_kernel(const float* __restrict__ g, float* __restrict__ h,
                            int N, int D, int H, int W, int Cout, int hidden,
                            int rev, int total)
{
    int i = blockIdx.x * blockDim.x + threadIdx.x;
    if (i >= total) return;                       // total = N*hidden*H*W
    size_t HW = (size_t)H * W;
    int x = i % W; int t = i / W;
    int y = t % H; t /= H;
    int c = t % hidden; int n = t / hidden;
    size_t sp = (size_t)y * W + x;
    size_t    d0   = rev ? (size_t)(D - 1) * HW : 0;
    ptrdiff_t step = rev ? -(ptrdiff_t)HW : (ptrdiff_t)HW;
    const float* zp = g + (((size_t)n * Cout + c) * D) * HW + sp + d0;
    const float* fp = g + (((size_t)n * Cout + c + hidden) * D) * HW + sp + d0;
    float*       hp = h + (((size_t)n * hidden + c) * D) * HW + sp + d0;
    float hv = 0.f;
    for (int s = 0; s < D; ++s) {
        float fv = *fp;
        hv = fv * hv + (1.f - fv) * (*zp);
        *hp = hv;
        zp += step; fp += step; hp += step;
    }
}

// ---------------------------------------------------------------------------

static inline int cdiv(int a, int b) { return (a + b - 1) / b; }

extern "C" void kernel_launch(void* const* d_in, const int* in_sizes, int n_in,
                              void* d_out, int out_size, void* d_ws, size_t ws_size,
                              hipStream_t stream)
{
    const float* x   = (const float*)d_in[0];
    const float* xs0 = (const float*)d_in[1];
    const float* xs1 = (const float*)d_in[2];
    const float* xs2 = (const float*)d_in[3];
    const float* w0  = (const float*)d_in[4];
    const float* w1  = (const float*)d_in[5];
    const float* w2  = (const float*)d_in[6];
    const float* w3  = (const float*)d_in[7];
    const float* bn0 = (const float*)d_in[8];
    const float* bn1 = (const float*)d_in[9];
    const float* bn2 = (const float*)d_in[10];
    const float* bn3 = (const float*)d_in[11];

    const int Nb = 4, Dd = 31;

    // ---- workspace layout ----
    char* ws = (char*)d_ws;
    bf16* WF0 = (bf16*)(ws);                       // 4 ob * 2 cb * 27 * 512
    bf16* WF1 = WF0 + (size_t)4 * 2 * 27 * 512;    // 4 * 1 * 27 * 512
    bf16* WF2 = WF1 + (size_t)4 * 1 * 27 * 512;    // 2 * 1 * 27 * 512
    bf16* WF3 = WF2 + (size_t)2 * 1 * 27 * 512;    // 2 * 1 * 27 * 512
    bf16*  BIN = (bf16*)(ws + (size_t)1  * 1024 * 1024);   // <= 143 MB
    float* G   = (float*)(ws + (size_t)161 * 1024 * 1024); // <= 260 MB
    float* Hb  = (float*)(ws + (size_t)425 * 1024 * 1024); // <= 130 MB
    float* out = (float*)d_out;

    // ---- weight prep (all 4 layers) ----
    { int tot = 4 * 2 * 27 * 32;
      wprep_kernel<<<cdiv(tot, 256), 256, 0, stream>>>(w0, WF0, 64, 64, 2, 0, tot); }
    { int tot = 4 * 1 * 27 * 32;
      wprep_kernel<<<cdiv(tot, 256), 256, 0, stream>>>(w1, WF1, 64, 32, 1, 1, tot); }
    { int tot = 2 * 1 * 27 * 32;
      wprep_kernel<<<cdiv(tot, 256), 256, 0, stream>>>(w2, WF2, 32, 32, 1, 0, tot); }
    { int tot = 2 * 1 * 27 * 32;
      wprep_kernel<<<cdiv(tot, 256), 256, 0, stream>>>(w3, WF3, 32, 16, 1, 1, tot); }

    // ---- layer 0: upsample(x) -> conv(w0) -> bn0 -> scan fwd, hidden 32 ----
    { const int H = 64, W = 64;
      int tp = Nb * (Dd + 2) * (H + 2) * (W + 2);
      inprep_kernel<<<cdiv(tp, 256), 256, 0, stream>>>(x, nullptr, BIN, Nb, 64, 64,
                                                       Dd, H, W, 1, tp);
      int waves = Nb * Dd * H * (W / 16) * (64 / 32);
      conv_wmma_kernel<31, 64, 64, 64, 32, 2>
          <<<cdiv(waves, 8), 256, 0, stream>>>(BIN, WF0, bn0, G, Nb, waves);
      int ts = Nb * 32 * H * W;
      scan_kernel<<<cdiv(ts, 256), 256, 0, stream>>>(G, Hb, Nb, Dd, H, W, 64, 32, 0, ts); }

    // ---- layer 1: (h + xs2) -> deconv(w1) -> bn1 -> scan rev, hidden 32 ----
    { const int H = 64, W = 64;
      int tp = Nb * (Dd + 2) * (H + 2) * (W + 2);
      inprep_kernel<<<cdiv(tp, 256), 256, 0, stream>>>(Hb, xs2, BIN, Nb, 32, 32,
                                                       Dd, H, W, 0, tp);
      int waves = Nb * Dd * H * (W / 16) * (64 / 32);
      conv_wmma_kernel<31, 64, 64, 64, 32, 1>
          <<<cdiv(waves, 8), 256, 0, stream>>>(BIN, WF1, bn1, G, Nb, waves);
      int ts = Nb * 32 * H * W;
      scan_kernel<<<cdiv(ts, 256), 256, 0, stream>>>(G, Hb, Nb, Dd, H, W, 64, 32, 1, ts); }

    // ---- layer 2: upsample(h + xs1) -> conv(w2) -> bn2 -> scan fwd, hidden 16 ----
    { const int H = 128, W = 128;
      int tp = Nb * (Dd + 2) * (H + 2) * (W + 2);
      inprep_kernel<<<cdiv(tp, 256), 256, 0, stream>>>(Hb, xs1, BIN, Nb, 32, 32,
                                                       Dd, H, W, 1, tp);
      int waves = Nb * Dd * H * (W / 16) * (32 / 32);
      conv_wmma_kernel<31, 128, 128, 32, 16, 1>
          <<<cdiv(waves, 8), 256, 0, stream>>>(BIN, WF2, bn2, G, Nb, waves);
      int ts = Nb * 16 * H * W;
      scan_kernel<<<cdiv(ts, 256), 256, 0, stream>>>(G, Hb, Nb, Dd, H, W, 32, 16, 0, ts); }

    // ---- layer 3: (h + xs0) -> deconv(w3) -> bn3 -> scan rev, hidden 16 ----
    { const int H = 128, W = 128;
      int tp = Nb * (Dd + 2) * (H + 2) * (W + 2);
      inprep_kernel<<<cdiv(tp, 256), 256, 0, stream>>>(Hb, xs0, BIN, Nb, 16, 32,
                                                       Dd, H, W, 0, tp);
      int waves = Nb * Dd * H * (W / 16) * (32 / 32);
      conv_wmma_kernel<31, 128, 128, 32, 16, 1>
          <<<cdiv(waves, 8), 256, 0, stream>>>(BIN, WF3, bn3, G, Nb, waves);
      int ts = Nb * 16 * H * W;
      scan_kernel<<<cdiv(ts, 256), 256, 0, stream>>>(G, out, Nb, Dd, H, W, 32, 16, 1, ts); }
}
